// MyCell_54262616818203
// MI455X (gfx1250) — matrix operand order repaired and checked by reading, not compiled, not run on hardware
//
#include <hip/hip_runtime.h>

// Problem sizes (fixed by the reference).
constexpr int kB = 32;
constexpr int kT = 2048;
constexpr int kD = 512;
constexpr int kU = 512;
// Chunked scan parameters: kT = kC * kNC.
constexpr int kC  = 32;   // chunk length (serial steps in phases A/C)
constexpr int kNC = 64;   // number of chunks (serial steps in phase B)

typedef float  v8f  __attribute__((ext_vector_type(8)));
typedef __bf16 v8bf __attribute__((ext_vector_type(8)));
typedef __bf16 v16bf __attribute__((ext_vector_type(16)));

__device__ __forceinline__ unsigned short f32_bf16(float f) {
    union { __bf16 h; unsigned short s; } u;
    u.h = (__bf16)f;            // native cvt (RTE)
    return u.s;
}

__device__ __forceinline__ v16bf cat16(v8bf lo, v8bf hi) {
    return __builtin_shufflevector(lo, hi, 0, 1, 2, 3, 4, 5, 6, 7,
                                           8, 9, 10, 11, 12, 13, 14, 15);
}

__device__ __forceinline__ v8f wmma_bf16(v16bf a, v16bf b, v8f c) {
    // D = A(16x32 bf16) * B(32x16 bf16) + C(16x16 f32)
    return __builtin_amdgcn_wmma_f32_16x16x32_bf16(
        false, a, false, b, (short)0, c, false, false);
}

// Load a wave's register-resident B-fragments of a 512x512 bf16 matrix.
// Wave owns N columns [wave*32, wave*32+32): 2 tiles x 16 K-slices.
__device__ __forceinline__ void load_Bfrags(const unsigned short* __restrict__ Rbf,
                                            int wave, int lane, v16bf (&Rreg)[2][16]) {
    const int ln = lane & 15;
    const int kb = (lane < 16) ? 0 : 16;
    #pragma unroll
    for (int j = 0; j < 2; ++j) {
        const int n = wave * 32 + j * 16 + ln;
        #pragma unroll
        for (int k = 0; k < 16; ++k) {
            union { unsigned short s[16]; v16bf v; } u;
            #pragma unroll
            for (int q = 0; q < 16; ++q)
                u.s[q] = Rbf[(size_t)(k * 32 + kb + q) * kU + n];
            Rreg[j][k] = u.v;
        }
    }
}

// acc[2] += h(16x512, LDS bf16 row stride 520) @ Rreg (wave's 32 columns).
__device__ __forceinline__ void h_times_R(const unsigned short* hbufp, int lane,
                                          const v16bf (&Rreg)[2][16], v8f (&acc)[2]) {
    const int ln = lane & 15;
    const int ak = (lane < 16) ? 0 : 16;  // contiguous k chunk base (k = 16h + s)
    const __bf16* hb = (const __bf16*)hbufp;
    #pragma unroll
    for (int k = 0; k < 16; ++k) {
        v8bf alo = *(const v8bf*)(hb + ln * 520 + k * 32 + ak);
        v8bf ahi = *(const v8bf*)(hb + ln * 520 + k * 32 + ak + 8);
        v16bf a  = cat16(alo, ahi);
        acc[0] = wmma_bf16(a, Rreg[0][k], acc[0]);
        acc[1] = wmma_bf16(a, Rreg[1][k], acc[1]);
    }
}

// ---------------------------------------------------------------------------
// Kernel 1: fp32 -> bf16 conversion of W and R.
// ---------------------------------------------------------------------------
__global__ void cvt_weights(const float* __restrict__ W, const float* __restrict__ R,
                            unsigned short* __restrict__ Wbf, unsigned short* __restrict__ Rbf) {
    int i = blockIdx.x * blockDim.x + threadIdx.x;
    if (i < kD * kU) {
        Wbf[i] = f32_bf16(W[i]);
        Rbf[i] = f32_bf16(R[i]);
    }
}

// ---------------------------------------------------------------------------
// Kernel 2: xk = x @ W  ([B*T,512]x[512,512] -> d_out fp32).
// ---------------------------------------------------------------------------
__launch_bounds__(128)
__global__ void gemm_xk(const float* __restrict__ x,
                        const unsigned short* __restrict__ Wbf,
                        float* __restrict__ out) {
    __shared__ __align__(16) unsigned short At[64 * 40];
    __shared__ __align__(16) unsigned short Bt[64 * 40];

    const int tid  = threadIdx.x;
    const int lane = tid & 31;
    const int wave = tid >> 5;
    const int ln   = lane & 15;
    const int m0   = blockIdx.y * 64;
    const int n0   = blockIdx.x * 64;

    v8f acc[4] = {};

    for (int ks = 0; ks < kD / 32; ++ks) {
        const int k0 = ks * 32;
        #pragma unroll
        for (int i = 0; i < 4; ++i) {
            const int row = (tid >> 3) + i * 16;
            const int col = (tid & 7) * 4;
            const float4 v = *(const float4*)(x + (size_t)(m0 + row) * kD + k0 + col);
            if (ks < kD / 32 - 1)   // global_prefetch_b8 of next K slice
                __builtin_prefetch(x + (size_t)(m0 + row) * kD + k0 + 32 + col, 0, 3);
            At[row * 40 + col + 0] = f32_bf16(v.x);
            At[row * 40 + col + 1] = f32_bf16(v.y);
            At[row * 40 + col + 2] = f32_bf16(v.z);
            At[row * 40 + col + 3] = f32_bf16(v.w);
        }
        {
            const int k  = tid >> 2;
            const int nc = (tid & 3) * 16;
            const unsigned short* src = Wbf + (size_t)(k0 + k) * kU + n0 + nc;
            #pragma unroll
            for (int i = 0; i < 16; ++i)
                Bt[(nc + i) * 40 + k] = src[i];
        }
        __syncthreads();

        const __bf16* Ab = (const __bf16*)At;
        const __bf16* Bb = (const __bf16*)Bt;
        const int arow = wave * 16 + ln;
        const int ak   = (lane < 16) ? 0 : 16;
        v8bf alo = *(const v8bf*)(Ab + arow * 40 + ak);
        v8bf ahi = *(const v8bf*)(Ab + arow * 40 + ak + 8);
        v16bf a  = cat16(alo, ahi);

        const int bk = (lane < 16) ? 0 : 16;
        #pragma unroll
        for (int j = 0; j < 4; ++j) {
            const int nr = j * 16 + ln;
            v8bf blo = *(const v8bf*)(Bb + nr * 40 + bk);
            v8bf bhi = *(const v8bf*)(Bb + nr * 40 + bk + 8);
            acc[j] = wmma_bf16(a, cat16(blo, bhi), acc[j]);
        }
        __syncthreads();
    }

    const int mb = (lane < 16) ? 0 : 8;
    #pragma unroll
    for (int j = 0; j < 4; ++j) {
        const int n = n0 + j * 16 + ln;
        #pragma unroll
        for (int r = 0; r < 8; ++r)
            out[(size_t)(m0 + wave * 16 + mb + r) * kU + n] = acc[j][r];
    }
}

// ---------------------------------------------------------------------------
// Kernel 3: bf16 512x512x512 GEMM (repeated squaring of R -> R^kC).
// A tile staged with CDNA5 async global->LDS DMA (pure byte copy, no
// convert/transpose needed), drained with s_wait_asynccnt before the barrier.
// ---------------------------------------------------------------------------
__launch_bounds__(128)
__global__ void gemm_pow(const unsigned short* __restrict__ A,
                         const unsigned short* __restrict__ B,
                         unsigned short* __restrict__ Cout) {
    __shared__ __align__(16) unsigned short At[64 * 40];
    __shared__ __align__(16) unsigned short Bt[64 * 40];

    const int tid  = threadIdx.x;
    const int lane = tid & 31;
    const int wave = tid >> 5;
    const int ln   = lane & 15;
    const int m0   = blockIdx.y * 64;
    const int n0   = blockIdx.x * 64;

    v8f acc[4] = {};

    // Per-thread A-stage mapping: thread -> (row, 32B half of 64B row slice).
    const int arow_st = tid >> 1;          // 0..63
    const int ahalf   = (tid & 1) * 16;    // element offset 0 / 16 (16 u16 = 32B)
    const unsigned int lds_at =
        (unsigned int)(size_t)&At[arow_st * 40 + ahalf];

    for (int ks = 0; ks < kD / 32; ++ks) {
        const int k0 = ks * 32;
        {   // async DMA: 2 x b128 per thread = 64 rows x 64 B
            const unsigned long long ga =
                (unsigned long long)(size_t)(A + (size_t)(m0 + arow_st) * kD + k0 + ahalf);
            asm volatile("global_load_async_to_lds_b128 %0, %1, off"
                         :: "v"(lds_at), "v"(ga) : "memory");
            asm volatile("global_load_async_to_lds_b128 %0, %1, off offset:16"
                         :: "v"(lds_at + 16u), "v"(ga) : "memory");
        }
        {
            const int k  = tid >> 2;
            const int nc = (tid & 3) * 16;
            const unsigned short* src = B + (size_t)(k0 + k) * kU + n0 + nc;
            #pragma unroll
            for (int i = 0; i < 16; ++i)
                Bt[(nc + i) * 40 + k] = src[i];
        }
        asm volatile("s_wait_asynccnt 0x0" ::: "memory");
        __syncthreads();

        const __bf16* Ab = (const __bf16*)At;
        const __bf16* Bb = (const __bf16*)Bt;
        const int arow = wave * 16 + ln;
        const int ak   = (lane < 16) ? 0 : 16;
        v8bf alo = *(const v8bf*)(Ab + arow * 40 + ak);
        v8bf ahi = *(const v8bf*)(Ab + arow * 40 + ak + 8);
        v16bf a  = cat16(alo, ahi);

        const int bk = (lane < 16) ? 0 : 16;
        #pragma unroll
        for (int j = 0; j < 4; ++j) {
            const int nr = j * 16 + ln;
            v8bf blo = *(const v8bf*)(Bb + nr * 40 + bk);
            v8bf bhi = *(const v8bf*)(Bb + nr * 40 + bk + 8);
            acc[j] = wmma_bf16(a, cat16(blo, bhi), acc[j]);
        }
        __syncthreads();
    }

    const int mb = (lane < 16) ? 0 : 8;
    #pragma unroll
    for (int j = 0; j < 4; ++j) {
        const int n = n0 + j * 16 + ln;
        #pragma unroll
        for (int r = 0; r < 8; ++r)
            Cout[(size_t)(m0 + wave * 16 + mb + r) * kU + n] = f32_bf16(acc[j][r]);
    }
}

// ---------------------------------------------------------------------------
// Phase A: local scan per chunk (zero entry state), in place over xk.
// Software-pipelined: prefetch xk[t+1] before the WMMA chain; defer the fp32
// store of h[t] to after the barrier. grid = (kNC, 2), 16 waves.
// ---------------------------------------------------------------------------
__launch_bounds__(512)
__global__ void scan_local(const unsigned short* __restrict__ Rbf,
                           float* __restrict__ out) {
    __shared__ __align__(16) unsigned short hbuf[2][16 * 520];

    const int tid  = threadIdx.x;
    const int lane = tid & 31;
    const int wave = tid >> 5;
    const int ln   = lane & 15;
    const int b0   = blockIdx.y * 16;
    const int t0   = blockIdx.x * kC;
    const int mb   = (lane < 16) ? 0 : 8;

    auto idx = [&](int t, int j, int r) -> size_t {
        const int n = wave * 32 + j * 16 + ln;
        return (size_t)(b0 + mb + r) * kT * kU + (size_t)t * kU + n;
    };

    v16bf Rreg[2][16];
    load_Bfrags(Rbf, wave, lane, Rreg);

    for (int i = tid; i < 16 * 520; i += 512) hbuf[0][i] = 0;
    __syncthreads();

    v8f cur[2], prev[2];
    #pragma unroll
    for (int j = 0; j < 2; ++j)
        #pragma unroll
        for (int r = 0; r < 8; ++r)
            cur[j][r] = out[idx(t0, j, r)];

    int p = 0;
    for (int tt = 0; tt < kC; ++tt) {
        const int t = t0 + tt;
        if (tt > 0) {            // deferred fp32 store of h[t-1]
            #pragma unroll
            for (int j = 0; j < 2; ++j)
                #pragma unroll
                for (int r = 0; r < 8; ++r)
                    out[idx(t - 1, j, r)] = prev[j][r];
        }
        v8f nxt[2] = {};
        if (tt + 1 < kC) {       // prefetch next xk (overlaps WMMA chain)
            #pragma unroll
            for (int j = 0; j < 2; ++j)
                #pragma unroll
                for (int r = 0; r < 8; ++r)
                    nxt[j][r] = out[idx(t + 1, j, r)];
        }

        v8f acc[2] = { cur[0], cur[1] };
        h_times_R(&hbuf[p][0], lane, Rreg, acc);

        unsigned short* ho = &hbuf[p ^ 1][0];
        #pragma unroll
        for (int j = 0; j < 2; ++j) {
            const int n = wave * 32 + j * 16 + ln;
            #pragma unroll
            for (int r = 0; r < 8; ++r)
                ho[(mb + r) * 520 + n] = f32_bf16(acc[j][r]);
        }
        __syncthreads();
        prev[0] = acc[0]; prev[1] = acc[1];
        cur[0]  = nxt[0]; cur[1]  = nxt[1];
        p ^= 1;
    }
    #pragma unroll
    for (int j = 0; j < 2; ++j)
        #pragma unroll
        for (int r = 0; r < 8; ++r)
            out[idx(t0 + kC - 1, j, r)] = prev[j][r];
}

// ---------------------------------------------------------------------------
// Phase B: boundary scan H_j = L_j + H_{j-1} @ R^kC over kNC chunks. grid=(2).
// ---------------------------------------------------------------------------
__launch_bounds__(512)
__global__ void scan_boundary(const unsigned short* __restrict__ Rpow,
                              const float* __restrict__ out,
                              unsigned short* __restrict__ Hbf) {
    __shared__ __align__(16) unsigned short hbuf[2][16 * 520];

    const int tid  = threadIdx.x;
    const int lane = tid & 31;
    const int wave = tid >> 5;
    const int ln   = lane & 15;
    const int half = blockIdx.x;
    const int b0   = half * 16;
    const int mb   = (lane < 16) ? 0 : 8;

    auto idxL = [&](int j, int jj, int r) -> size_t {   // L_j read (chunk-end row)
        const int t = j * kC + (kC - 1);
        const int n = wave * 32 + jj * 16 + ln;
        return (size_t)(b0 + mb + r) * kT * kU + (size_t)t * kU + n;
    };
    auto idxH = [&](int j, int jj, int r) -> size_t {   // H_j write
        const int n = wave * 32 + jj * 16 + ln;
        return ((size_t)(half * kNC + j) * 16 + mb + r) * kU + n;
    };

    v16bf Rreg[2][16];
    load_Bfrags(Rpow, wave, lane, Rreg);

    for (int i = tid; i < 16 * 520; i += 512) hbuf[0][i] = 0;
    __syncthreads();

    v8f cur[2], prev[2];
    #pragma unroll
    for (int jj = 0; jj < 2; ++jj)
        #pragma unroll
        for (int r = 0; r < 8; ++r)
            cur[jj][r] = out[idxL(0, jj, r)];

    int p = 0;
    for (int j = 0; j < kNC; ++j) {
        if (j > 0) {             // deferred bf16 store of H_{j-1}
            #pragma unroll
            for (int jj = 0; jj < 2; ++jj)
                #pragma unroll
                for (int r = 0; r < 8; ++r)
                    Hbf[idxH(j - 1, jj, r)] = f32_bf16(prev[jj][r]);
        }
        v8f nxt[2] = {};
        if (j + 1 < kNC) {
            #pragma unroll
            for (int jj = 0; jj < 2; ++jj)
                #pragma unroll
                for (int r = 0; r < 8; ++r)
                    nxt[jj][r] = out[idxL(j + 1, jj, r)];
        }

        v8f acc[2] = { cur[0], cur[1] };
        h_times_R(&hbuf[p][0], lane, Rreg, acc);

        unsigned short* ho = &hbuf[p ^ 1][0];
        #pragma unroll
        for (int jj = 0; jj < 2; ++jj) {
            const int n = wave * 32 + jj * 16 + ln;
            #pragma unroll
            for (int r = 0; r < 8; ++r)
                ho[(mb + r) * 520 + n] = f32_bf16(acc[jj][r]);
        }
        __syncthreads();
        prev[0] = acc[0]; prev[1] = acc[1];
        cur[0]  = nxt[0]; cur[1]  = nxt[1];
        p ^= 1;
    }
    #pragma unroll
    for (int jj = 0; jj < 2; ++jj)
        #pragma unroll
        for (int r = 0; r < 8; ++r)
            Hbf[idxH(kNC - 1, jj, r)] = f32_bf16(prev[jj][r]);
}

// ---------------------------------------------------------------------------
// Phase C: correction for chunk j>=1: corr_t = corr_{t-1} @ R seeded with
// H_{j-1}; out[t] += corr_t.  grid = (kNC-1, 2).
// ---------------------------------------------------------------------------
__launch_bounds__(512)
__global__ void scan_correct(const unsigned short* __restrict__ Rbf,
                             float* __restrict__ out,
                             const unsigned short* __restrict__ Hbf) {
    __shared__ __align__(16) unsigned short hbuf[2][16 * 520];

    const int tid  = threadIdx.x;
    const int lane = tid & 31;
    const int wave = tid >> 5;
    const int ln   = lane & 15;
    const int j    = blockIdx.x + 1;
    const int half = blockIdx.y;
    const int b0   = half * 16;
    const int mb   = (lane < 16) ? 0 : 8;

    auto idx = [&](int t, int jj, int r) -> size_t {
        const int n = wave * 32 + jj * 16 + ln;
        return (size_t)(b0 + mb + r) * kT * kU + (size_t)t * kU + n;
    };

    v16bf Rreg[2][16];
    load_Bfrags(Rbf, wave, lane, Rreg);

    // seed corr state = H_{j-1} (bf16)
    for (int i = tid; i < 16 * 512; i += 512) {
        const int m = i >> 9, n = i & 511;
        hbuf[0][m * 520 + n] = Hbf[((size_t)(half * kNC + j - 1) * 16 + m) * kU + n];
    }
    __syncthreads();

    const int t0 = j * kC;
    v8f cur[2], prev[2];
    #pragma unroll
    for (int jj = 0; jj < 2; ++jj)
        #pragma unroll
        for (int r = 0; r < 8; ++r)
            cur[jj][r] = out[idx(t0, jj, r)];

    int p = 0;
    for (int tt = 0; tt < kC; ++tt) {
        const int t = t0 + tt;
        if (tt > 0) {            // deferred store of corrected h[t-1]
            #pragma unroll
            for (int jj = 0; jj < 2; ++jj)
                #pragma unroll
                for (int r = 0; r < 8; ++r)
                    out[idx(t - 1, jj, r)] = prev[jj][r];
        }
        v8f nxt[2] = {};
        if (tt + 1 < kC) {
            #pragma unroll
            for (int jj = 0; jj < 2; ++jj)
                #pragma unroll
                for (int r = 0; r < 8; ++r)
                    nxt[jj][r] = out[idx(t + 1, jj, r)];
        }

        v8f acc[2] = {};         // corr_t = corr_{t-1} @ R
        h_times_R(&hbuf[p][0], lane, Rreg, acc);

        unsigned short* ho = &hbuf[p ^ 1][0];
        #pragma unroll
        for (int jj = 0; jj < 2; ++jj) {
            const int n = wave * 32 + jj * 16 + ln;
            #pragma unroll
            for (int r = 0; r < 8; ++r)
                ho[(mb + r) * 520 + n] = f32_bf16(acc[jj][r]);
        }
        v8f sum[2] = { cur[0] + acc[0], cur[1] + acc[1] };
        __syncthreads();
        prev[0] = sum[0]; prev[1] = sum[1];
        cur[0]  = nxt[0]; cur[1]  = nxt[1];
        p ^= 1;
    }
    #pragma unroll
    for (int jj = 0; jj < 2; ++jj)
        #pragma unroll
        for (int r = 0; r < 8; ++r)
            out[idx(t0 + kC - 1, jj, r)] = prev[jj][r];
}

// ---------------------------------------------------------------------------
extern "C" void kernel_launch(void* const* d_in, const int* in_sizes, int n_in,
                              void* d_out, int out_size, void* d_ws, size_t ws_size,
                              hipStream_t stream) {
    const float* x = (const float*)d_in[0];
    const float* W = (const float*)d_in[1];  // kernel [D, U]
    const float* R = (const float*)d_in[2];  // recurrent_kernel [U, U]
    float* out = (float*)d_out;              // [B, T, U]

    unsigned short* Wbf = (unsigned short*)d_ws;         // 512 KB
    unsigned short* Rbf = Wbf + (size_t)kD * kU;         // 512 KB
    unsigned short* pa  = Rbf + (size_t)kD * kU;         // 512 KB (power ping)
    unsigned short* pb  = pa  + (size_t)kD * kU;         // 512 KB (power pong)
    unsigned short* Hbf = pb  + (size_t)kD * kU;         // 2 MB: [2][kNC][16][512] bf16

    hipLaunchKernelGGL(cvt_weights, dim3((kD * kU + 255) / 256), dim3(256), 0, stream,
                       W, R, Wbf, Rbf);
    hipLaunchKernelGGL(gemm_xk, dim3(kU / 64, (kB * kT) / 64), dim3(128), 0, stream,
                       x, Wbf, out);

    // R^32 by repeated squaring (bf16): R^2, R^4, R^8, R^16, R^32
    hipLaunchKernelGGL(gemm_pow, dim3(8, 8), dim3(128), 0, stream, Rbf, Rbf, pa);
    hipLaunchKernelGGL(gemm_pow, dim3(8, 8), dim3(128), 0, stream, pa, pa, pb);
    hipLaunchKernelGGL(gemm_pow, dim3(8, 8), dim3(128), 0, stream, pb, pb, pa);
    hipLaunchKernelGGL(gemm_pow, dim3(8, 8), dim3(128), 0, stream, pa, pa, pb);
    hipLaunchKernelGGL(gemm_pow, dim3(8, 8), dim3(128), 0, stream, pb, pb, pa);

    hipLaunchKernelGGL(scan_local,    dim3(kNC, 2),     dim3(512), 0, stream, Rbf, out);
    hipLaunchKernelGGL(scan_boundary, dim3(2),          dim3(512), 0, stream, pa, out, Hbf);
    hipLaunchKernelGGL(scan_correct,  dim3(kNC - 1, 2), dim3(512), 0, stream, Rbf, out, Hbf);
}
